// MultiHeadModulator_25409026523859
// MI455X (gfx1250) — matrix operand order repaired and checked by reading, not compiled, not run on hardware
//
#include <hip/hip_runtime.h>
#include <hip/hip_bf16.h>
#include <cstdint>

// ---------------------------------------------------------------------------
// MultiHeadModulator for MI455X (gfx1250, wave32, WMMA + async-to-LDS)
//   DIM=512, HEADS=8, B=16, REL_MAX=64, RANK=4, ALPHA=16, QBITS=4, d2=1024
// ---------------------------------------------------------------------------

#define D2      1024      // 2*DIM
#define BSZ     16
#define HEADS_N 8
#define HD2     128       // head dim * 2 (interleaved re/im)
#define QMAX    7.0f      // 2^(4-1)-1
#define LORA_S  4.0f      // ALPHA / RANK

typedef __attribute__((ext_vector_type(16))) _Float16 v16h;
typedef __attribute__((ext_vector_type(8)))  float    v8f;
typedef __attribute__((ext_vector_type(4)))  float    v4f;
typedef __attribute__((ext_vector_type(4)))  unsigned v4u;
typedef __attribute__((ext_vector_type(4)))  int      v4i;

#if __has_builtin(__builtin_amdgcn_global_load_async_to_lds_b128) && \
    __has_builtin(__builtin_amdgcn_s_wait_asynccnt)
#define USE_ASYNC_LDS 1
#else
#define USE_ASYNC_LDS 0
#endif

// builtin parameter types: (AS1 v4i*, AS3 v4i*, imm offset, imm cpol)
typedef __attribute__((address_space(1))) v4i g_v4i;
typedef __attribute__((address_space(3))) v4i l_v4i;

// ---------------------------------------------------------------------------
// Fragment loaders with contiguous k per lane half (A and B use the same order)
// A (16x32 f16, MxK): lane<16 -> M=lane,   K = 0..15
//                     lane>=16-> M=lane-16,K = 16..31
// ---------------------------------------------------------------------------
__device__ __forceinline__ v16h load_a_f32(const float* __restrict__ X, int ld,
                                           int m, int k0, int lane) {
  int kb = k0 + ((lane & 16) ? 16 : 0);
  const float* p = X + (size_t)m * ld + kb;
  v4f x0 = *(const v4f*)(p + 0);     // k+0..3
  v4f x1 = *(const v4f*)(p + 4);     // k+4..7
  v4f x2 = *(const v4f*)(p + 8);     // k+8..11
  v4f x3 = *(const v4f*)(p + 12);    // k+12..15
  v16h a;
  a[0]  = (_Float16)x0[0]; a[1]  = (_Float16)x0[1];
  a[2]  = (_Float16)x0[2]; a[3]  = (_Float16)x0[3];
  a[4]  = (_Float16)x1[0]; a[5]  = (_Float16)x1[1];
  a[6]  = (_Float16)x1[2]; a[7]  = (_Float16)x1[3];
  a[8]  = (_Float16)x2[0]; a[9]  = (_Float16)x2[1];
  a[10] = (_Float16)x2[2]; a[11] = (_Float16)x2[3];
  a[12] = (_Float16)x3[0]; a[13] = (_Float16)x3[1];
  a[14] = (_Float16)x3[2]; a[15] = (_Float16)x3[3];
  return a;
}

// B (32x16 f16, KxN) from row-major [n][k] storage (B[k][n] = W[n][k]):
//   lane<16 -> N=lane, K=k0+0..15 ; lane>=16 -> N=lane-16, K=k0+16..31
__device__ __forceinline__ v16h load_b_f16(const _Float16* __restrict__ W, int ld,
                                           int n, int k0, int lane) {
  int k = k0 + ((lane >> 4) << 4);
  return *(const v16h*)(W + (size_t)n * ld + k);
}

// ---------------------------------------------------------------------------
// 0) zero the absmax slots
// ---------------------------------------------------------------------------
__global__ void init_slots(unsigned* slots) {
  if (threadIdx.x < 8) slots[threadIdx.x] = 0u;
}

// ---------------------------------------------------------------------------
// 1) absmax over each W (slots 0..3) and each bias (slots 4..7)
// ---------------------------------------------------------------------------
__global__ __launch_bounds__(256) void absmax_kernel(
    const float* __restrict__ W0, const float* __restrict__ W1,
    const float* __restrict__ W2, const float* __restrict__ W3,
    const float* __restrict__ b0, const float* __restrict__ b1,
    const float* __restrict__ b2, const float* __restrict__ b3,
    unsigned* __restrict__ slots) {
  __shared__ float red[256];
  int which = blockIdx.y;
  const float* src; int n;
  switch (which) {
    case 0: src = W0; n = D2 * D2; break;
    case 1: src = W1; n = D2 * D2; break;
    case 2: src = W2; n = D2 * D2; break;
    case 3: src = W3; n = D2 * D2; break;
    case 4: src = b0; n = D2; break;
    case 5: src = b1; n = D2; break;
    case 6: src = b2; n = D2; break;
    default: src = b3; n = D2; break;
  }
  int t = blockIdx.x * blockDim.x + threadIdx.x;
  int stride = gridDim.x * blockDim.x;
  float m = 0.0f;
  for (int i = t; i < n; i += stride) m = fmaxf(m, fabsf(src[i]));
  red[threadIdx.x] = m;
  __syncthreads();
  for (int s = 128; s > 0; s >>= 1) {
    if (threadIdx.x < s) red[threadIdx.x] = fmaxf(red[threadIdx.x], red[threadIdx.x + s]);
    __syncthreads();
  }
  if (threadIdx.x == 0)
    atomicMax(&slots[which], __float_as_uint(red[0]));   // valid: values >= 0
}

// ---------------------------------------------------------------------------
// 2) W_eff[n][k] = simq(W)[n][k] + (ALPHA/RANK) * sum_r LB[n][r]*A[r][k], f16
// ---------------------------------------------------------------------------
__global__ __launch_bounds__(256) void build_weff(
    const float* __restrict__ W0, const float* __restrict__ W1,
    const float* __restrict__ W2, const float* __restrict__ W3,
    const float* __restrict__ A0, const float* __restrict__ A1,
    const float* __restrict__ A2, const float* __restrict__ A3,
    const float* __restrict__ L0, const float* __restrict__ L1,
    const float* __restrict__ L2, const float* __restrict__ L3,
    const unsigned* __restrict__ slots,
    _Float16* __restrict__ E0, _Float16* __restrict__ E1,
    _Float16* __restrict__ E2, _Float16* __restrict__ E3) {
  int w = blockIdx.y;
  const float *W, *A, *L; _Float16* E;
  switch (w) {
    case 0: W = W0; A = A0; L = L0; E = E0; break;
    case 1: W = W1; A = A1; L = L1; E = E1; break;
    case 2: W = W2; A = A2; L = L2; E = E2; break;
    default: W = W3; A = A3; L = L3; E = E3; break;
  }
  int idx = blockIdx.x * blockDim.x + threadIdx.x;
  if (idx >= D2 * D2) return;
  int n = idx >> 10;          // /1024
  int k = idx & (D2 - 1);
  float mabs  = __uint_as_float(slots[w]);
  float scale = mabs / QMAX;
  float wv = W[idx];
  float q;
  if (mabs == 0.0f) {
    q = wv;
  } else {
    q = fminf(fmaxf(rintf(wv / (scale + 1e-8f)), -8.0f), 7.0f) * scale;
  }
  float delta = 0.0f;
  #pragma unroll
  for (int r = 0; r < 4; ++r) delta += L[n * 4 + r] * A[r * D2 + k];
  E[idx] = (_Float16)(q + LORA_S * delta);
}

// ---------------------------------------------------------------------------
// 3) beff[w][n] = simq(b_w)[n]
// ---------------------------------------------------------------------------
__global__ __launch_bounds__(256) void build_beff(
    const float* __restrict__ b0, const float* __restrict__ b1,
    const float* __restrict__ b2, const float* __restrict__ b3,
    const unsigned* __restrict__ slots, float* __restrict__ beff) {
  int idx = blockIdx.x * blockDim.x + threadIdx.x;
  if (idx >= 4 * D2) return;
  int w = idx >> 10;
  int n = idx & (D2 - 1);
  const float* b = (w == 0) ? b0 : (w == 1) ? b1 : (w == 2) ? b2 : b3;
  float mabs  = __uint_as_float(slots[4 + w]);
  float scale = mabs / QMAX;
  float bv = b[n];
  float q;
  if (mabs == 0.0f) q = bv;
  else q = fminf(fmaxf(rintf(bv / (scale + 1e-8f)), -8.0f), 7.0f) * scale;
  beff[idx] = q;
}

// ---------------------------------------------------------------------------
// 4) Fused K + V projection GEMM via WMMA with double-buffered async B tiles.
//    Out[m][n] = sum_k X[m][k] * Weff[n][k] + bias[n] for both Wk and Wv.
//    Block = 256 thr (8 waves). Tile: M=128 (wave -> 16 rows), N=64, Kstage=64.
//    LDS: Bbuf[stage][weight][64 n][64 k + 16 pad halves] = 40 KB.
// ---------------------------------------------------------------------------
#define KSTAGE 64
#define BPAD   80   // 64 k halves + 16 pad -> 160B row (32B aligned, 40-bank stride)

__global__ __launch_bounds__(256) void kv_gemm(
    const float* __restrict__ X,        // [Mrows][1024] f32 (z_past real view)
    const _Float16* __restrict__ Wk, const _Float16* __restrict__ Wv,
    const float* __restrict__ bk, const float* __restrict__ bv,
    _Float16* __restrict__ Kout, _Float16* __restrict__ Vout) {
  __shared__ __align__(32) _Float16 Bbuf[2][2][64][BPAD];

  int tid  = threadIdx.x;
  int wave = tid >> 5;
  int lane = tid & 31;
  int m0 = blockIdx.y * 128 + wave * 16;
  int n0 = blockIdx.x * 64;
  int mA = m0 + (lane & 15);            // A-fragment row for this lane
  int nl = lane & 15;
  int kh = (lane >> 4) << 4;            // 0 or 16 (B fragment k-half)

  // Stage one 64-wide K slab of both weight matrices into LDS.
  // 1024 chunks of 16B; chunk c = i*256 + tid (lane-contiguous -> coalesced).
  auto stage_load = [&](int s, int k0) {
    #pragma unroll
    for (int i = 0; i < 4; ++i) {
      int c = i * 256 + tid;
      int w = c >> 9;                   // 0: Wk, 1: Wv
      int rem = c & 511;
      int r = rem >> 3;                 // n_local 0..63
      int q = rem & 7;                  // 16B chunk within 128B k-row
      const _Float16* g = (w ? Wv : Wk) + (size_t)(n0 + r) * D2 + k0 + q * 8;
      _Float16* l = &Bbuf[s][w][r][q * 8];
#if USE_ASYNC_LDS
      __builtin_amdgcn_global_load_async_to_lds_b128(
          (g_v4i*)(uintptr_t)g, (l_v4i*)(v4i*)l, 0, 0);
#else
      *(v4u*)l = *(const v4u*)g;
#endif
    }
  };

  v8f accK[4] = {v8f{}, v8f{}, v8f{}, v8f{}};
  v8f accV[4] = {v8f{}, v8f{}, v8f{}, v8f{}};

  stage_load(0, 0);
#if USE_ASYNC_LDS
  __builtin_amdgcn_s_wait_asynccnt(0);
#endif
  __syncthreads();

  int s = 0;
  for (int k0 = 0; k0 < D2; k0 += KSTAGE, s ^= 1) {
    if (k0 + KSTAGE < D2) {
      stage_load(s ^ 1, k0 + KSTAGE);                      // overlap with WMMA
      __builtin_prefetch(X + (size_t)mA * D2 + k0 + KSTAGE, 0, 3);
    }
    #pragma unroll
    for (int ks = 0; ks < 2; ++ks) {
      v16h a = load_a_f32(X, D2, mA, k0 + ks * 32, lane);

      // batch the K-weight fragments: ds_loads issue as a clause, WMMAs
      // retire against staggered dscnt waits instead of 0-wait each.
      v16h bK0 = *(const v16h*)&Bbuf[s][0][ 0 + nl][ks * 32 + kh];
      v16h bK1 = *(const v16h*)&Bbuf[s][0][16 + nl][ks * 32 + kh];
      v16h bK2 = *(const v16h*)&Bbuf[s][0][32 + nl][ks * 32 + kh];
      v16h bK3 = *(const v16h*)&Bbuf[s][0][48 + nl][ks * 32 + kh];
      accK[0] = __builtin_amdgcn_wmma_f32_16x16x32_f16(
          false, a, false, bK0, (short)0, accK[0], false, false);
      accK[1] = __builtin_amdgcn_wmma_f32_16x16x32_f16(
          false, a, false, bK1, (short)0, accK[1], false, false);
      accK[2] = __builtin_amdgcn_wmma_f32_16x16x32_f16(
          false, a, false, bK2, (short)0, accK[2], false, false);
      accK[3] = __builtin_amdgcn_wmma_f32_16x16x32_f16(
          false, a, false, bK3, (short)0, accK[3], false, false);

      v16h bV0 = *(const v16h*)&Bbuf[s][1][ 0 + nl][ks * 32 + kh];
      v16h bV1 = *(const v16h*)&Bbuf[s][1][16 + nl][ks * 32 + kh];
      v16h bV2 = *(const v16h*)&Bbuf[s][1][32 + nl][ks * 32 + kh];
      v16h bV3 = *(const v16h*)&Bbuf[s][1][48 + nl][ks * 32 + kh];
      accV[0] = __builtin_amdgcn_wmma_f32_16x16x32_f16(
          false, a, false, bV0, (short)0, accV[0], false, false);
      accV[1] = __builtin_amdgcn_wmma_f32_16x16x32_f16(
          false, a, false, bV1, (short)0, accV[1], false, false);
      accV[2] = __builtin_amdgcn_wmma_f32_16x16x32_f16(
          false, a, false, bV2, (short)0, accV[2], false, false);
      accV[3] = __builtin_amdgcn_wmma_f32_16x16x32_f16(
          false, a, false, bV3, (short)0, accV[3], false, false);
    }
#if USE_ASYNC_LDS
    __builtin_amdgcn_s_wait_asynccnt(0);
#endif
    __syncthreads();
  }

  // C/D layout: VGPR r, lane l -> M = r + 8*(l>=16), N = l%16
  int mh = (lane >> 4) * 8;
  #pragma unroll
  for (int t = 0; t < 4; ++t) {
    int n = n0 + t * 16 + nl;
    float bbK = bk[n];
    float bbV = bv[n];
    #pragma unroll
    for (int r = 0; r < 8; ++r) {
      size_t row = (size_t)(m0 + mh + r);
      Kout[row * D2 + n] = (_Float16)(accK[t][r] + bbK);
      Vout[row * D2 + n] = (_Float16)(accV[t][r] + bbV);
    }
  }
}

// ---------------------------------------------------------------------------
// 5) M=16 projection (Q and O paths): one wave per 16x16 output tile.
// ---------------------------------------------------------------------------
__global__ __launch_bounds__(32) void proj16(
    const float* __restrict__ Xin,      // [16][1024] f32
    const _Float16* __restrict__ W,     // [1024][1024] f16
    const float* __restrict__ bias,     // [1024]
    float* __restrict__ out) {          // [16][1024] f32
  int lane = threadIdx.x;
  int n0 = blockIdx.x * 16;
  int mA = lane & 15;
  v8f acc = v8f{};
  for (int k0 = 0; k0 < D2; k0 += 32) {
    v16h a = load_a_f32(Xin, D2, mA, k0, lane);
    v16h b = load_b_f16(W, D2, n0 + (lane & 15), k0, lane);
    acc = __builtin_amdgcn_wmma_f32_16x16x32_f16(
        false, a, false, b, (short)0, acc, false, false);
  }
  int n = n0 + (lane & 15);
  int mh = (lane >> 4) * 8;
  float bb = bias[n];
  #pragma unroll
  for (int r = 0; r < 8; ++r)
    out[(size_t)(mh + r) * D2 + n] = acc[r] + bb;
}

// ---------------------------------------------------------------------------
// 6) Attention: per (b,h): scores over P, softmax, weighted V -> [16][1024]
//    Re(q conj(k)) over head == plain dot over the interleaved 128 values.
// ---------------------------------------------------------------------------
__global__ __launch_bounds__(256) void attn_kernel(
    const float* __restrict__ qbuf,       // [16][1024] f32
    const _Float16* __restrict__ Kp,      // [P*16][1024] f16
    const _Float16* __restrict__ Vp,      // [P*16][1024] f16
    const float* __restrict__ relb,       // [REL_SIZE][1024] f32
    const int* __restrict__ curr_pos_p,
    float* __restrict__ attn_out,         // [16][1024] f32
    int P, int rel_size) {
  __shared__ float sq[HD2];
  __shared__ float sw[2048];
  __shared__ float red[256];
  int bh = blockIdx.x;
  int b = bh >> 3;
  int h = bh & 7;
  int tid = threadIdx.x;

  if (tid < HD2) sq[tid] = qbuf[(size_t)b * D2 + h * HD2 + tid];
  __syncthreads();

  int curr = curr_pos_p[0];
  for (int p = tid; p < P; p += 256) {
    int rel_d = curr - P + p;
    int idx = rel_d + 64;
    idx = (idx < 0) ? 0 : ((idx > rel_size - 1) ? rel_size - 1 : idx);
    const _Float16* kp = Kp + ((size_t)p * BSZ + b) * D2 + h * HD2;
    const float* rp = relb + (size_t)idx * D2 + h * HD2;
    float s = 0.0f;
    #pragma unroll 4
    for (int j = 0; j < HD2; ++j)
      s += sq[j] * ((float)kp[j] + rp[j]);
    sw[p] = s * 0.125f;                  // 1/sqrt(hd) = 1/8
  }
  __syncthreads();

  float mx = -3.4e38f;
  for (int p = tid; p < P; p += 256) mx = fmaxf(mx, sw[p]);
  red[tid] = mx;
  __syncthreads();
  for (int s = 128; s > 0; s >>= 1) {
    if (tid < s) red[tid] = fmaxf(red[tid], red[tid + s]);
    __syncthreads();
  }
  float rowmax = red[0];
  __syncthreads();

  float sum = 0.0f;
  for (int p = tid; p < P; p += 256) {
    float e = __expf(sw[p] - rowmax);
    sw[p] = e;
    sum += e;
  }
  red[tid] = sum;
  __syncthreads();
  for (int s = 128; s > 0; s >>= 1) {
    if (tid < s) red[tid] += red[tid + s];
    __syncthreads();
  }
  float inv = 1.0f / red[0];
  __syncthreads();

  int j = tid & 127;
  int half = tid >> 7;
  int pstart = half * (P >> 1);
  int pend = pstart + (P >> 1);
  float acc = 0.0f;
  for (int p = pstart; p < pend; ++p)
    acc += sw[p] * (float)Vp[((size_t)p * BSZ + b) * D2 + h * HD2 + j];
  red[tid] = acc * inv;
  __syncthreads();
  if (tid < HD2)
    attn_out[(size_t)b * D2 + h * HD2 + tid] = red[tid] + red[tid + HD2];
}

// ---------------------------------------------------------------------------
// launch
// ---------------------------------------------------------------------------
extern "C" void kernel_launch(void* const* d_in, const int* in_sizes, int n_in,
                              void* d_out, int out_size, void* d_ws, size_t ws_size,
                              hipStream_t stream) {
  const int*   curr_pos = (const int*)d_in[0];
  const float* z_curr   = (const float*)d_in[1];   // [16][1024] view
  const float* z_past   = (const float*)d_in[2];   // [P*16][1024] view
  const float* relb     = (const float*)d_in[3];   // [REL_SIZE][1024] view

  const float *W[4], *bs[4], *A[4], *LB[4];
  for (int i = 0; i < 4; ++i) {
    W[i]  = (const float*)d_in[4 + 4 * i];
    bs[i] = (const float*)d_in[5 + 4 * i];
    A[i]  = (const float*)d_in[6 + 4 * i];
    LB[i] = (const float*)d_in[7 + 4 * i];
  }
  int P        = in_sizes[2] / (BSZ * D2);   // 2048
  int rel_size = in_sizes[3] / D2;           // 129

  // workspace layout (all segment sizes multiples of 256B)
  char* ws = (char*)d_ws;
  unsigned* slots = (unsigned*)ws;                                   // 8 u32
  size_t off = 256;
  _Float16* Weff[4];
  for (int w = 0; w < 4; ++w) { Weff[w] = (_Float16*)(ws + off); off += (size_t)D2 * D2 * 2; }
  float* beff = (float*)(ws + off);     off += 4 * D2 * 4;           // [4][1024]
  float* qbuf = (float*)(ws + off);     off += (size_t)BSZ * D2 * 4; // [16][1024]
  float* aout = (float*)(ws + off);     off += (size_t)BSZ * D2 * 4; // [16][1024]
  _Float16* Kp = (_Float16*)(ws + off); off += (size_t)P * BSZ * D2 * 2;
  _Float16* Vp = (_Float16*)(ws + off); off += (size_t)P * BSZ * D2 * 2;

  init_slots<<<1, 32, 0, stream>>>(slots);

  absmax_kernel<<<dim3(256, 8), 256, 0, stream>>>(
      W[0], W[1], W[2], W[3], bs[0], bs[1], bs[2], bs[3], slots);

  build_weff<<<dim3((D2 * D2 + 255) / 256, 4), 256, 0, stream>>>(
      W[0], W[1], W[2], W[3],
      A[0], A[1], A[2], A[3],
      LB[0], LB[1], LB[2], LB[3],
      slots, Weff[0], Weff[1], Weff[2], Weff[3]);

  build_beff<<<(4 * D2 + 255) / 256, 256, 0, stream>>>(
      bs[0], bs[1], bs[2], bs[3], slots, beff);

  kv_gemm<<<dim3(D2 / 64, (P * BSZ) / 128), 256, 0, stream>>>(
      z_past, Weff[1], Weff[2], beff + D2, beff + 2 * D2, Kp, Vp);

  proj16<<<D2 / 16, 32, 0, stream>>>(z_curr, Weff[0], beff, qbuf);

  attn_kernel<<<BSZ * HEADS_N, 256, 0, stream>>>(
      qbuf, Kp, Vp, relb, curr_pos, aout, P, rel_size);

  proj16<<<D2 / 16, 32, 0, stream>>>(aout, Weff[3], beff + 3 * D2, (float*)d_out);
}
